// BiLSTM_11647951307115
// MI455X (gfx1250) — compile-verified, matrix operand order checked by reading, not run on hardware
//
#include <hip/hip_runtime.h>
#include <hip/hip_bf16.h>

// BiLSTM for MI455X (gfx1250), wave32 + WMMA bf16.
//
//  1) pack x into WMMA A-layout bf16 tiles (natural + per-sample-reversed copies)
//  2) pack Wx_f/Wx_b/Wh_f/Wh_b into WMMA B-layout bf16 tiles
//  3) fused bias vectors  bias = bx + bh  (folded into xproj)
//  4) xproj GEMM per direction: [16384,512]x[512,2048], 4 N-tiles/wave sharing A
//  5) persistent recurrence: 16 WGs per direction (cluster-sized). Each WG owns
//     hidden slice [wg*32, wg*32+32): 8 N-tiles whose Wh panel (128 KB bf16)
//     is LDS-resident for all 16384 steps. h is exchanged through L2 with
//     device-scope atomics (64 B per WG per step) + atomic-counter barrier
//     (+ s_cluster_barrier, NOP when not cluster-dispatched). c stays in regs.

#define Bsz 32
#define Tsz 512
#define Esz 512
#define Hsz 512
#define G4H 2048            // 4*H
#define SEQ (Bsz * Tsz)     // 16384
#define KT  (Esz / 32)      // 16 k-steps of 32 per WMMA chain
#define MT  (SEQ / 16)      // 1024 M-tiles
#define NT  (G4H / 16)      // 128 N-tiles
#define TILE_ELEMS (KT * 32 * 16)   // 8192 bf16 per 16x512 operand panel
#define NWG 16              // workgroups per direction in the recurrence

typedef __bf16 bf16_t;
typedef __attribute__((ext_vector_type(16))) __bf16 v16bf;
typedef __attribute__((ext_vector_type(8)))  float  v8f;

// ISA 16-bit A/B lane layout (CDNA5 ISA §7.12.2):
// lanes 0-15 hold K = {slot 0..7 -> 0..7,  slot 8..15 -> 16..23}
// lanes 16-31 hold K = {slot 0..7 -> 8..15, slot 8..15 -> 24..31}
__device__ __forceinline__ int koff_map(int lane, int slot) {
    int base = (slot < 8) ? slot : slot + 8;
    return (lane < 16) ? base : base + 8;
}

__device__ __forceinline__ float sigm(float x) {
    return 1.0f / (1.0f + __expf(-x));
}

// ---------------- packing kernels ----------------

// x [SEQ, E] f32 -> A-layout bf16 tiles [mt][kt][lane][slot]
__global__ void __launch_bounds__(256)
pack_x_kernel(const float* __restrict__ x, bf16_t* __restrict__ dst, int reverse) {
    int idx  = blockIdx.x * 256 + threadIdx.x;       // total MT*KT*32*16 = 8388608
    int slot = idx & 15;
    int lane = (idx >> 4) & 31;
    int kt   = (idx >> 9) & 15;
    int mt   = idx >> 13;
    int k    = kt * 32 + koff_map(lane, slot);
    int row  = mt * 16 + (lane & 15);
    if (reverse) { int b = row >> 9, t = row & 511; row = (b << 9) + (511 - t); }
    dst[idx] = (bf16_t)x[(size_t)row * Esz + k];
}

// W [4H, K=512] f32 -> B-layout bf16 tiles [nt][kt][lane][slot]
__global__ void __launch_bounds__(256)
pack_w_kernel(const float* __restrict__ W, bf16_t* __restrict__ dst) {
    int idx  = blockIdx.x * 256 + threadIdx.x;       // total NT*KT*32*16 = 1048576
    int slot = idx & 15;
    int lane = (idx >> 4) & 31;
    int kt   = (idx >> 9) & 15;
    int nt   = idx >> 13;
    int k    = kt * 32 + koff_map(lane, slot);
    int n    = nt * 16 + (lane & 15);
    dst[idx] = (bf16_t)W[(size_t)n * 512 + k];
}

__global__ void __launch_bounds__(256)
bias_kernel(const float* bxf, const float* bhf, const float* bxb, const float* bhb,
            float* biasF, float* biasB) {
    int i = blockIdx.x * 256 + threadIdx.x;
    if (i < G4H) {
        biasF[i] = bxf[i] + bhf[i];
        biasB[i] = bxb[i] + bhb[i];
    }
}

// zero the h-exchange buffer (2 dirs x 256 dwords) and the 2 barrier counters
__global__ void __launch_bounds__(256)
init_sync_kernel(unsigned* hbuf, unsigned* bar) {
    int i = blockIdx.x * 256 + threadIdx.x;
    if (i < 2 * 256) hbuf[i] = 0u;
    if (i < 2)       bar[i]  = 0u;
}

// ---------------- xproj GEMM (WMMA bf16) ----------------
// block = 8 waves; each wave computes 4 adjacent 16x16 N-tiles sharing one A panel.
__global__ void __launch_bounds__(256)
gemm_xproj(const bf16_t* __restrict__ Apack, const bf16_t* __restrict__ Bpack,
           const float* __restrict__ bias, float* __restrict__ Cout) {
    const int lane = threadIdx.x & 31;
    const int wave = threadIdx.x >> 5;
    const int mt   = blockIdx.x;                        // 0..MT-1
    const int ntb  = (blockIdx.y * 8 + wave) * 4;       // 0,4,...,124

    const bf16_t* aptr = Apack + (size_t)mt * TILE_ELEMS + (size_t)lane * 16;
    const bf16_t* b0   = Bpack + (size_t)(ntb + 0) * TILE_ELEMS + (size_t)lane * 16;
    const bf16_t* b1   = Bpack + (size_t)(ntb + 1) * TILE_ELEMS + (size_t)lane * 16;
    const bf16_t* b2   = Bpack + (size_t)(ntb + 2) * TILE_ELEMS + (size_t)lane * 16;
    const bf16_t* b3   = Bpack + (size_t)(ntb + 3) * TILE_ELEMS + (size_t)lane * 16;

    v8f acc0 = {}, acc1 = {}, acc2 = {}, acc3 = {};
#pragma unroll
    for (int kt = 0; kt < KT; ++kt) {
        size_t ko = (size_t)kt * (32 * 16);
        __builtin_prefetch(aptr + ko + 1024, 0, 1);     // global_prefetch_b8
        v16bf a  = *(const v16bf*)(aptr + ko);
        v16bf f0 = *(const v16bf*)(b0 + ko);
        v16bf f1 = *(const v16bf*)(b1 + ko);
        v16bf f2 = *(const v16bf*)(b2 + ko);
        v16bf f3 = *(const v16bf*)(b3 + ko);
        acc0 = __builtin_amdgcn_wmma_f32_16x16x32_bf16(false, a, false, f0, (short)0, acc0, false, false);
        acc1 = __builtin_amdgcn_wmma_f32_16x16x32_bf16(false, a, false, f1, (short)0, acc1, false, false);
        acc2 = __builtin_amdgcn_wmma_f32_16x16x32_bf16(false, a, false, f2, (short)0, acc2, false, false);
        acc3 = __builtin_amdgcn_wmma_f32_16x16x32_bf16(false, a, false, f3, (short)0, acc3, false, false);
    }

    const int rowbase = mt * 16 + ((lane < 16) ? 0 : 8);
    const int c       = lane & 15;
#pragma unroll
    for (int t = 0; t < 4; ++t) {
        const int col = (ntb + t) * 16 + c;
        const float bv = bias[col];
        const v8f acc = (t == 0) ? acc0 : (t == 1) ? acc1 : (t == 2) ? acc2 : acc3;
#pragma unroll
        for (int i = 0; i < 8; ++i) {
            Cout[(size_t)(rowbase + i) * G4H + col] = acc[i] + bv;
        }
    }
}

// ---------------- persistent recurrence, LDS-resident weights ----------------
// grid = (NWG, 2); block = 256 (8 waves). Wave w handles N-tile
// nt = (w>>1)*32 + wg*2 + (w&1)  (gate g = w>>1, half = w&1), i.e. the WG owns
// gate columns {g*512 + wg*32 + [0,32)} for all four gates -> hidden slice
// j in [wg*32, wg*32+32). h is exchanged via hbuf (global, A-pack order:
// fragment kt == wg is exactly this WG's 16 dwords).
__global__ void __launch_bounds__(256)
bilstm_scan(const float* __restrict__ xprojF, const float* __restrict__ xprojB,
            const bf16_t* __restrict__ WhF, const bf16_t* __restrict__ WhB,
            unsigned* __restrict__ hbuf, unsigned* __restrict__ bar,
            float* __restrict__ out) {
    const int wg  = blockIdx.x;       // 0..15
    const int dir = blockIdx.y;       // 0 fwd, 1 bwd
    const float*  xp = dir ? xprojB : xprojF;
    const bf16_t* Wh = dir ? WhB : WhF;
    unsigned* hb  = hbuf + dir * 256; // 256 dwords = 512 bf16, A-pack order
    unsigned* bcv = bar + dir;

    extern __shared__ __attribute__((aligned(32))) char smem[];
    bf16_t* wh_lds = (bf16_t*)smem;                      // 8 tiles * 8192 = 128 KB
    bf16_t* h_lds  = (bf16_t*)(smem + 131072);           // 512 bf16 (A-pack order)
    float*  g_sh   = (float*)(smem + 131072 + 1024);     // 128 gate preacts
    bf16_t* h_st   = (bf16_t*)(smem + 131072 + 1024 + 512); // 32 new h values

    const int tid  = threadIdx.x;
    const int lane = tid & 31;
    const int wave = tid >> 5;
    const int gate = wave >> 1;
    const int half = wave & 1;
    const int nt   = gate * 32 + wg * 2 + half;

    // stage this WG's Wh panel into LDS once (8 tiles of 16 KB)
    const bf16_t* wsrc = Wh + (size_t)nt * TILE_ELEMS + (size_t)lane * 16;
    bf16_t*       wdst = wh_lds + wave * TILE_ELEMS + lane * 16;
#pragma unroll
    for (int kt = 0; kt < KT; ++kt) {
        *(v16bf*)(wdst + kt * 512) = *(const v16bf*)(wsrc + (size_t)kt * 512);
    }
    __syncthreads();

    float creg = 0.0f;                                   // cell state, in-register
    const int colg = nt * 16 + (lane & 15);              // this lane's gate column

    for (int step = 0; step < SEQ; ++step) {
        // gather full h (device-scope: produced by sibling WGPs last step)
        if (tid < 256) {
            ((unsigned*)h_lds)[tid] =
                __hip_atomic_load(hb + tid, __ATOMIC_RELAXED, __HIP_MEMORY_SCOPE_AGENT);
        }
        if (step + 1 < SEQ)   // warm next step's xproj slice in GL2
            __builtin_prefetch(xp + (size_t)(step + 1) * G4H + colg, 0, 1);
        __syncthreads();

        // h @ Wh^T for this wave's 16x16 tile; B operands stream from LDS
        v8f acc = {};
#pragma unroll
        for (int kt = 0; kt < KT; ++kt) {
            v16bf a = {};
            if ((lane & 15) == 0)
                a = *(const v16bf*)(h_lds + kt * 32 + (lane & 16));
            v16bf b = *(const v16bf*)(wh_lds + wave * TILE_ELEMS + kt * 512 + lane * 16);
            acc = __builtin_amdgcn_wmma_f32_16x16x32_bf16(false, a, false, b,
                                                          (short)0, acc, false, false);
        }

        // D row 0 (lanes 0..15, VGPR0) + xproj (+ folded biases)
        if (lane < 16) {
            g_sh[gate * 32 + half * 16 + lane] = acc[0] + xp[(size_t)step * G4H + colg];
        }
        __syncthreads();

        // cell update for the 32 owned hidden units; c stays in registers
        if (tid < 32) {
            const int lh = tid, j = wg * 32 + lh;
            float f  = sigm(g_sh[0 * 32 + lh]);
            float ii = sigm(g_sh[1 * 32 + lh]);
            float o  = sigm(g_sh[2 * 32 + lh]);
            float gt = tanhf(g_sh[3 * 32 + lh]);
            creg = f * creg + ii * gt;
            float h = o * tanhf(creg);
            h_st[lh] = (bf16_t)h;

            size_t oidx;
            if (dir == 0) {
                oidx = (size_t)step * (2 * Hsz) + j;                   // [b,t, 0:H]
            } else {
                int b = step >> 9, t = 511 - (step & 511);             // un-reverse
                oidx = ((size_t)(b * Tsz + t)) * (2 * Hsz) + Hsz + j;  // [b,t, H:2H]
            }
            out[oidx] = h;
        }
        __syncthreads();

        // publish fragment kt==wg of the packed h (16 dwords, device scope)
        if (tid < 16) {
            const int d  = tid;
            const int hf = d >> 3;
            const int s0 = (d & 7) * 2;                       // even slot
            const int k0 = (hf == 0) ? (s0 < 8 ? s0 : s0 + 8)
                                     : (s0 < 8 ? s0 + 8 : s0 + 16);
            unsigned lo = __builtin_bit_cast(unsigned short, h_st[k0]);
            unsigned hi = __builtin_bit_cast(unsigned short, h_st[k0 + 1]);
            __hip_atomic_store(hb + wg * 16 + d, lo | (hi << 16),
                               __ATOMIC_RELAXED, __HIP_MEMORY_SCOPE_AGENT);
        }
        __syncthreads();

        // inter-WG barrier: release our arrival, then wait for all NWG
        if (tid == 0)
            __hip_atomic_fetch_add(bcv, 1u, __ATOMIC_RELEASE, __HIP_MEMORY_SCOPE_AGENT);
        __builtin_amdgcn_s_cluster_barrier();   // real barrier iff cluster-dispatched
        if (tid == 0) {
            const unsigned target = (unsigned)NWG * (unsigned)(step + 1);
            while (__hip_atomic_load(bcv, __ATOMIC_ACQUIRE,
                                     __HIP_MEMORY_SCOPE_AGENT) < target)
                __builtin_amdgcn_s_sleep(8);
        }
        __syncthreads();
    }
}

// ---------------- host side ----------------

extern "C" void kernel_launch(void* const* d_in, const int* in_sizes, int n_in,
                              void* d_out, int out_size, void* d_ws, size_t ws_size,
                              hipStream_t stream) {
    (void)in_sizes; (void)n_in; (void)out_size; (void)ws_size;

    const float* x    = (const float*)d_in[0];
    const float* Wx_f = (const float*)d_in[1];
    const float* bx_f = (const float*)d_in[2];
    const float* Wh_f = (const float*)d_in[3];
    const float* bh_f = (const float*)d_in[4];
    const float* Wx_b = (const float*)d_in[5];
    const float* bx_b = (const float*)d_in[6];
    const float* Wh_b = (const float*)d_in[7];
    const float* bh_b = (const float*)d_in[8];
    float* out = (float*)d_out;

    // workspace carve-up (bytes); total ~310 MB
    char* ws = (char*)d_ws;
    size_t off = 0;
    const size_t xpackB = (size_t)SEQ * Esz * sizeof(bf16_t);   // 16 MB
    const size_t wpackB = (size_t)G4H * 512 * sizeof(bf16_t);   //  2 MB
    const size_t xprojB = (size_t)SEQ * G4H * sizeof(float);    // 134 MB

    bf16_t* XpF = (bf16_t*)(ws + off); off += xpackB;
    bf16_t* XpB = (bf16_t*)(ws + off); off += xpackB;
    bf16_t* WxF = (bf16_t*)(ws + off); off += wpackB;
    bf16_t* WxB = (bf16_t*)(ws + off); off += wpackB;
    bf16_t* WhF = (bf16_t*)(ws + off); off += wpackB;
    bf16_t* WhB = (bf16_t*)(ws + off); off += wpackB;
    float*  bF  = (float*)(ws + off);  off += G4H * sizeof(float);
    float*  bB  = (float*)(ws + off);  off += G4H * sizeof(float);
    float*  xpF = (float*)(ws + off);  off += xprojB;
    float*  xpB = (float*)(ws + off);  off += xprojB;
    unsigned* hbuf = (unsigned*)(ws + off); off += 2 * 256 * sizeof(unsigned);
    unsigned* bar  = (unsigned*)(ws + off); off += 256;

    // 0) deterministic re-init of cross-WG sync state (graph-replay safe)
    init_sync_kernel<<<2, 256, 0, stream>>>(hbuf, bar);

    // 1) pack activations (forward + time-reversed-per-sample)
    const int xTot = MT * KT * 32 * 16;           // 8388608
    pack_x_kernel<<<xTot / 256, 256, 0, stream>>>(x, XpF, 0);
    pack_x_kernel<<<xTot / 256, 256, 0, stream>>>(x, XpB, 1);

    // 2) pack weights into WMMA B-layout
    const int wTot = NT * KT * 32 * 16;           // 1048576
    pack_w_kernel<<<wTot / 256, 256, 0, stream>>>(Wx_f, WxF);
    pack_w_kernel<<<wTot / 256, 256, 0, stream>>>(Wx_b, WxB);
    pack_w_kernel<<<wTot / 256, 256, 0, stream>>>(Wh_f, WhF);
    pack_w_kernel<<<wTot / 256, 256, 0, stream>>>(Wh_b, WhB);

    // 3) fused biases bx + bh
    bias_kernel<<<(G4H + 255) / 256, 256, 0, stream>>>(bx_f, bh_f, bx_b, bh_b, bF, bB);

    // 4) input projections (WMMA GEMM), 4 N-tiles per wave
    dim3 ggrid(MT, NT / 32);
    gemm_xproj<<<ggrid, 256, 0, stream>>>(XpF, WxF, bF, xpF);
    gemm_xproj<<<ggrid, 256, 0, stream>>>(XpB, WxB, bB, xpB);

    // 5) persistent recurrence: 16 WGs x 2 directions, Wh LDS-resident
    const size_t scan_lds = 131072 + 1024 + 512 + 64;   // ~130 KB per WG
    dim3 sgrid(NWG, 2);
    bilstm_scan<<<sgrid, 256, scan_lds, stream>>>(xpF, xpB, WhF, WhB, hbuf, bar, out);
}
